// MultiHeadedAttention_52999896432880
// MI455X (gfx1250) — matrix operand order repaired and checked by reading, not compiled, not run on hardware
//
#include <hip/hip_runtime.h>

// MHA: B=2, S=2048, E=1024, H=16, D=64. fp32 in/out, f16 WMMA internally.
// CDNA5 paths: v_wmma_f32_16x16x32_f16 + global_load_async_to_lds_b128
// (ASYNCcnt double-buffered LDS staging) + global_prefetch.
#define EMBED 1024
#define HEADS 16
#define HDIM  64
#define BATCH 2
#define SEQ   2048

typedef _Float16 v16h __attribute__((ext_vector_type(16)));
typedef float    v8f  __attribute__((ext_vector_type(8)));

static __device__ __forceinline__ v8f wmma_f16(v16h a, v16h b, v8f c) {
  // D = A(16x32,f16) * B(32x16,f16) + C(16x16,f32)
  return __builtin_amdgcn_wmma_f32_16x16x32_f16(false, a, false, b, (short)0, c,
                                                false, false);
}

static __device__ __forceinline__ v8f zero8() {
  v8f z = {0.f, 0.f, 0.f, 0.f, 0.f, 0.f, 0.f, 0.f};
  return z;
}

// Async copy 16 bytes/lane: global -> LDS, tracked by ASYNCcnt.
// lds_off is the wave-relative LDS byte address (low 32 bits of a generic
// pointer to __shared__), g is the per-lane global source address.
static __device__ __forceinline__ void async_copy16(unsigned lds_off,
                                                    const void* g) {
  asm volatile("global_load_async_to_lds_b128 %0, %1, off" ::"v"(lds_off),
               "v"(g)
               : "memory");
}
static __device__ __forceinline__ unsigned lds_addr(const void* p) {
  return (unsigned)(unsigned long long)p;
}
static __device__ __forceinline__ void wait_async_le2() {
  asm volatile("s_wait_asynccnt 0x2" ::: "memory");
}
static __device__ __forceinline__ void wait_async_0() {
  asm volatile("s_wait_asynccnt 0x0" ::: "memory");
}

// max/sum reduction across the 16-lane half-group (xor 1,2,4,8 stays in-group)
static __device__ __forceinline__ float rmax16(float x) {
#pragma unroll
  for (int off = 1; off < 16; off <<= 1) x = fmaxf(x, __shfl_xor(x, off, 32));
  return x;
}
static __device__ __forceinline__ float rsum16(float x) {
#pragma unroll
  for (int off = 1; off < 16; off <<= 1) x += __shfl_xor(x, off, 32);
  return x;
}

// ---------------------------------------------------------------------------
// Stage 1: fused QKV projection. out = X @ W^T + b  (Q scaled by 1/sqrt(D)).
// blockIdx.z: 0=Q, 1=K, 2=V. Q/K stored [B,H,S,D] f16, V stored [B,H,D,S] f16.
// 8 waves/block, wave = 16x64 tile. The shared 64x32 fp32 W tile is async-
// staged into LDS (double buffered); A-fragments stream straight from global.
// ---------------------------------------------------------------------------
__global__ void __launch_bounds__(256)
qkv_proj_kernel(const float* __restrict__ Xq, const float* __restrict__ Xk,
                const float* __restrict__ Xv, const float* __restrict__ Wq,
                const float* __restrict__ Wk, const float* __restrict__ Wv,
                const float* __restrict__ bq, const float* __restrict__ bk,
                const float* __restrict__ bv, _Float16* __restrict__ Qh,
                _Float16* __restrict__ Kh, _Float16* __restrict__ Vt) {
  __shared__ __align__(16) float wbuf[2][64][32];  // 2 x 8 KB weight tile

  const int mode = blockIdx.z;
  const float* X    = (mode == 0) ? Xq : (mode == 1) ? Xk : Xv;
  const float* W    = (mode == 0) ? Wq : (mode == 1) ? Wk : Wv;
  const float* bias = (mode == 0) ? bq : (mode == 1) ? bk : bv;

  const int tid  = threadIdx.x;
  const int wave = tid >> 5;
  const int lane = tid & 31;
  const int hl   = lane >> 4;
  const int l15  = lane & 15;
  const int m0   = blockIdx.x * 128 + wave * 16;  // row base (B*S = 4096)
  const int n0   = blockIdx.y * 64;               // col base (E = 1024)

  // issue async copy of W[n0..n0+63][k0..k0+31] into wbuf[s]
  auto issue_w = [&](int s, int k0) {
#pragma unroll
    for (int i = 0; i < 2; ++i) {
      const int c   = tid + 256 * i;   // 512 x 16B chunks
      const int row = c >> 3;
      const int col = (c & 7) * 4;
      async_copy16(lds_addr(&wbuf[s][row][col]),
                   W + (size_t)(n0 + row) * EMBED + k0 + col);
    }
  };

  v8f acc[4];
#pragma unroll
  for (int t = 0; t < 4; ++t) acc[t] = zero8();

  const float* arow = X + (size_t)(m0 + l15) * EMBED;

  issue_w(0, 0);
  for (int it = 0; it < EMBED / 32; ++it) {
    const int k0 = it * 32;
    if (it + 1 < EMBED / 32) {
      issue_w((it + 1) & 1, k0 + 32);
      wait_async_le2();  // current tile's 2 copies done; next 2 in flight
    } else {
      wait_async_0();
    }
    __syncthreads();

    // A-fragment (16x32): lane holds row l15, K = hl*16..+15
    v16h a;
    const float* pa = arow + k0 + hl * 16;
#pragma unroll
    for (int j = 0; j < 8; ++j) {
      a[j]     = (_Float16)pa[j];
      a[8 + j] = (_Float16)pa[8 + j];
    }
    const int s = it & 1;
#pragma unroll
    for (int t = 0; t < 4; ++t) {
      // B-fragment (32x16): lane = column (n0+t*16+l15), K = hl*16..+15
      const float* pb = &wbuf[s][t * 16 + l15][hl * 16];
      v16h bm;
#pragma unroll
      for (int j = 0; j < 16; ++j) bm[j] = (_Float16)pb[j];
      acc[t] = wmma_f16(a, bm, acc[t]);
    }
    __syncthreads();  // tile fully consumed before it is refilled
  }

#pragma unroll
  for (int t = 0; t < 4; ++t) {
    const int n      = n0 + t * 16 + l15;
    const float bval = bias[n];
    const int h = n >> 6, d = n & 63;
#pragma unroll
    for (int r = 0; r < 8; ++r) {
      const int m  = m0 + r + 8 * hl;
      const int bb = m >> 11, sdx = m & (SEQ - 1);
      float val = acc[t][r] + bval;
      if (mode == 0) val *= 0.125f;  // 1/sqrt(64)
      if (mode < 2) {
        _Float16* dst = (mode == 0) ? Qh : Kh;
        dst[((size_t)(bb * HEADS + h) * SEQ + sdx) * HDIM + d] = (_Float16)val;
      } else {
        Vt[((size_t)(bb * HEADS + h) * HDIM + d) * SEQ + sdx] = (_Float16)val;
      }
    }
  }
}

// ---------------------------------------------------------------------------
// Stage 2: flash attention. 8 waves/block share one (b,h); each wave owns a
// 16-query tile. Per 32-key chunk the block async-stages the shared K (32x64
// f16) and V^T (64x32 f16) tiles into double-buffered LDS, then each wave:
// 4 WMMA (QK^T) + online softmax + P relayout via LDS + 4 WMMA (P*V).
// ---------------------------------------------------------------------------
__global__ void __launch_bounds__(256)
attn_kernel(const _Float16* __restrict__ Qh, const _Float16* __restrict__ Kh,
            const _Float16* __restrict__ Vt, const int* __restrict__ mask,
            _Float16* __restrict__ ctx) {
  __shared__ __align__(32) _Float16 kbuf[2][32][64];  // 2 x 4 KB
  __shared__ __align__(32) _Float16 vbuf[2][64][32];  // 2 x 4 KB
  __shared__ __align__(32) _Float16 pbuf[8][16][32];  // 8 KB (per wave)

  const int tid  = threadIdx.x;
  const int wave = tid >> 5;
  const int lane = tid & 31;
  const int hl   = lane >> 4;
  const int l15  = lane & 15;
  const int b    = blockIdx.z;
  const int h    = blockIdx.y;
  const int q0   = blockIdx.x * 128 + wave * 16;
  const int bh   = b * HEADS + h;

  const _Float16* Qb = Qh + (size_t)bh * SEQ * HDIM;
  const _Float16* Kb = Kh + (size_t)bh * SEQ * HDIM;
  const _Float16* Vb = Vt + (size_t)bh * HDIM * SEQ;
  const int* Mb      = mask + (size_t)b * SEQ * SEQ;

  // issue async copies of this chunk's K and V tiles into buffers [s]
  auto issue_kv = [&](int s, int kc) {
    {  // K tile: 32 rows x 64 halves = 256 x 16B chunks
      const int row = tid >> 3, ch = (tid & 7) * 8;
      async_copy16(lds_addr(&kbuf[s][row][ch]),
                   Kb + (size_t)(kc + row) * HDIM + ch);
    }
    {  // V tile: 64 rows x 32 halves = 256 x 16B chunks
      const int row = tid >> 2, ch = (tid & 3) * 8;
      async_copy16(lds_addr(&vbuf[s][row][ch]),
                   Vb + (size_t)row * SEQ + kc + ch);
    }
  };

  // Q A-fragments for d = 0..31 and 32..63 (loaded once, straight from global)
  v16h qa[2];
#pragma unroll
  for (int t = 0; t < 2; ++t) {
    const _Float16* p = Qb + (size_t)(q0 + l15) * HDIM + t * 32 + hl * 16;
#pragma unroll
    for (int j = 0; j < 8; ++j) {
      qa[t][j]     = p[j];
      qa[t][8 + j] = p[8 + j];
    }
  }

  v8f o[4];
#pragma unroll
  for (int g = 0; g < 4; ++g) o[g] = zero8();
  v8f mst, lst;
#pragma unroll
  for (int r = 0; r < 8; ++r) {
    mst[r] = -3.0e38f;
    lst[r] = 0.0f;
  }

  issue_kv(0, 0);
  for (int it = 0; it < SEQ / 32; ++it) {
    const int kc = it * 32;
    if (it + 1 < SEQ / 32) {
      issue_kv((it + 1) & 1, kc + 32);
      // prefetch next chunk's mask (one 128B line per query row of this wave)
      __builtin_prefetch(Mb + (size_t)(q0 + l15) * SEQ + kc + 32, 0, 1);
      wait_async_le2();
    } else {
      wait_async_0();
    }
    __syncthreads();  // K/V tiles for this chunk visible block-wide
    const int s = it & 1;

    // ---- scores: two 16x16 tiles over this 32-key chunk ----
    v8f s0 = zero8(), s1 = zero8();
#pragma unroll
    for (int t = 0; t < 2; ++t) {
      v16h kb0 = *(const v16h*)&kbuf[s][l15][t * 32 + hl * 16];
      v16h kb1 = *(const v16h*)&kbuf[s][16 + l15][t * 32 + hl * 16];
      s0 = wmma_f16(qa[t], kb0, s0);
      s1 = wmma_f16(qa[t], kb1, s1);
    }
    // ---- mask: score = -10000 where mask==0 ----
#pragma unroll
    for (int r = 0; r < 8; ++r) {
      const int* mrow = Mb + (size_t)(q0 + r + 8 * hl) * SEQ + kc;
      s0[r] = (mrow[l15] == 0) ? -1.0e4f : s0[r];
      s1[r] = (mrow[16 + l15] == 0) ? -1.0e4f : s1[r];
    }
    // ---- online softmax (per query row, replicated across its 16 lanes) ----
    v8f corr;
#pragma unroll
    for (int r = 0; r < 8; ++r) {
      float mx   = rmax16(fmaxf(s0[r], s1[r]));
      float mnew = fmaxf(mst[r], mx);
      corr[r]    = __expf(mst[r] - mnew);
      mst[r]     = mnew;
      s0[r]      = __expf(s0[r] - mnew);
      s1[r]      = __expf(s1[r] - mnew);
      lst[r]     = lst[r] * corr[r] + rsum16(s0[r] + s1[r]);
    }
#pragma unroll
    for (int g = 0; g < 4; ++g)
#pragma unroll
      for (int r = 0; r < 8; ++r) o[g][r] *= corr[r];

    // ---- re-layout P: C-fragment -> LDS -> A-fragment ----
    __syncthreads();
#pragma unroll
    for (int r = 0; r < 8; ++r) {
      pbuf[wave][r + 8 * hl][l15]      = (_Float16)s0[r];
      pbuf[wave][r + 8 * hl][16 + l15] = (_Float16)s1[r];
    }
    __syncthreads();
    v16h pa;
#pragma unroll
    for (int j = 0; j < 8; ++j) {
      pa[j]     = pbuf[wave][l15][hl * 16 + j];
      pa[8 + j] = pbuf[wave][l15][hl * 16 + 8 + j];
    }
    // ---- O += P * V : B-fragment columns are output d (V staged as [D,k]) --
#pragma unroll
    for (int g = 0; g < 4; ++g) {
      v16h vb = *(const v16h*)&vbuf[s][g * 16 + l15][hl * 16];
      o[g] = wmma_f16(pa, vb, o[g]);
    }
    __syncthreads();  // K/V/P tiles consumed before next chunk's refill
  }

  // ---- normalize and write context [B,S,E] f16 ----
#pragma unroll
  for (int g = 0; g < 4; ++g)
#pragma unroll
    for (int r = 0; r < 8; ++r) {
      const int qrow = q0 + r + 8 * hl;
      ctx[((size_t)b * SEQ + qrow) * EMBED + h * HDIM + g * 16 + l15] =
          (_Float16)(o[g][r] / lst[r]);
    }
}

// ---------------------------------------------------------------------------
// Stage 3: output projection. out(fp32) = ctx(f16) @ Wo^T + bo.
// Same async-staged weight-tile structure as stage 1.
// ---------------------------------------------------------------------------
__global__ void __launch_bounds__(256)
out_proj_kernel(const _Float16* __restrict__ ctx, const float* __restrict__ Wo,
                const float* __restrict__ bo, float* __restrict__ out) {
  __shared__ __align__(16) float wbuf[2][64][32];

  const int tid  = threadIdx.x;
  const int wave = tid >> 5;
  const int lane = tid & 31;
  const int hl   = lane >> 4;
  const int l15  = lane & 15;
  const int m0   = blockIdx.x * 128 + wave * 16;
  const int n0   = blockIdx.y * 64;

  auto issue_w = [&](int s, int k0) {
#pragma unroll
    for (int i = 0; i < 2; ++i) {
      const int c   = tid + 256 * i;
      const int row = c >> 3;
      const int col = (c & 7) * 4;
      async_copy16(lds_addr(&wbuf[s][row][col]),
                   Wo + (size_t)(n0 + row) * EMBED + k0 + col);
    }
  };

  v8f acc[4];
#pragma unroll
  for (int t = 0; t < 4; ++t) acc[t] = zero8();

  const _Float16* arow = ctx + (size_t)(m0 + l15) * EMBED;

  issue_w(0, 0);
  for (int it = 0; it < EMBED / 32; ++it) {
    const int k0 = it * 32;
    if (it + 1 < EMBED / 32) {
      issue_w((it + 1) & 1, k0 + 32);
      wait_async_le2();
    } else {
      wait_async_0();
    }
    __syncthreads();

    v16h a;
    const _Float16* pa = arow + k0 + hl * 16;
#pragma unroll
    for (int j = 0; j < 8; ++j) {
      a[j]     = pa[j];
      a[8 + j] = pa[8 + j];
    }
    const int s = it & 1;
#pragma unroll
    for (int t = 0; t < 4; ++t) {
      const float* pb = &wbuf[s][t * 16 + l15][hl * 16];
      v16h bm;
#pragma unroll
      for (int j = 0; j < 16; ++j) bm[j] = (_Float16)pb[j];
      acc[t] = wmma_f16(a, bm, acc[t]);
    }
    __syncthreads();
  }

#pragma unroll
  for (int t = 0; t < 4; ++t) {
    const int n      = n0 + t * 16 + l15;
    const float bval = bo[n];
#pragma unroll
    for (int r = 0; r < 8; ++r) {
      const int m = m0 + r + 8 * hl;
      out[(size_t)m * EMBED + n] = acc[t][r] + bval;
    }
  }
}

// ---------------------------------------------------------------------------
extern "C" void kernel_launch(void* const* d_in, const int* in_sizes, int n_in,
                              void* d_out, int out_size, void* d_ws,
                              size_t ws_size, hipStream_t stream) {
  const float* q   = (const float*)d_in[0];
  const float* k   = (const float*)d_in[1];
  const float* v   = (const float*)d_in[2];
  const int* mask  = (const int*)d_in[3];
  const float* Wq  = (const float*)d_in[4];
  const float* bq  = (const float*)d_in[5];
  const float* Wk  = (const float*)d_in[6];
  const float* bk  = (const float*)d_in[7];
  const float* Wv  = (const float*)d_in[8];
  const float* bv  = (const float*)d_in[9];
  const float* Wo  = (const float*)d_in[10];
  const float* bo  = (const float*)d_in[11];

  const size_t QKV = (size_t)BATCH * HEADS * SEQ * HDIM;  // 4M halves each
  _Float16* Qh  = (_Float16*)d_ws;
  _Float16* Kh  = Qh + QKV;
  _Float16* Vt  = Kh + QKV;
  _Float16* ctx = Vt + QKV;  // [B,S,E] f16 — total 32 MB workspace

  qkv_proj_kernel<<<dim3(32, 16, 3), 256, 0, stream>>>(
      q, k, v, Wq, Wk, Wv, bq, bk, bv, Qh, Kh, Vt);
  attn_kernel<<<dim3(SEQ / 128, HEADS, BATCH), 256, 0, stream>>>(
      Qh, Kh, Vt, mask, ctx);
  out_proj_kernel<<<dim3(32, 16, 1), 256, 0, stream>>>(
      ctx, Wo, bo, (float*)d_out);
}
